// GPT2_32822140076271
// MI455X (gfx1250) — matrix operand order repaired and checked by reading, not compiled, not run on hardware
//
#include <hip/hip_runtime.h>
#include <hip/hip_bf16.h>

// ---------------- model constants ----------------
#define BB   8
#define SS   1024
#define DATA 64
#define DD   512
#define HH   8
#define DHH  64
#define DFF  2048
#define LL   4
#define MM   (BB * SS)          // 8192 rows of activations

typedef __attribute__((ext_vector_type(16))) __bf16 v16bf;
typedef __attribute__((ext_vector_type(8)))  float  v8f;

#define BF16_NEG1E9 0xCE6Eu    // f2bf(-1e9f)

union ABf {
  v16bf v;
  unsigned short u[16];
};

__device__ __forceinline__ unsigned short f2bf(float f) {
  unsigned int u = __float_as_uint(f);
  u += 0x7FFFu + ((u >> 16) & 1u);        // round-to-nearest-even
  return (unsigned short)(u >> 16);
}

__device__ __forceinline__ float gelu_exact(float x) {
  return 0.5f * x * (1.0f + erff(x * 0.7071067811865475f));
}

// CDNA5 async global->LDS byte copy (16B per lane), tracked by ASYNCcnt.
__device__ __forceinline__ void async_copy_b128(unsigned lds_off, const void* gaddr) {
  asm volatile("global_load_async_to_lds_b128 %0, %1, off"
               :: "v"(lds_off), "v"(gaddr) : "memory");
}
__device__ __forceinline__ void wait_async0() {
  asm volatile("s_wait_asynccnt 0x0" ::: "memory");
}

// ---------------- positional encoding ----------------
__global__ void pe_kernel(float* __restrict__ pe) {
  int idx = blockIdx.x * blockDim.x + threadIdx.x;   // over S*D
  int s = idx / DD, d = idx % DD;
  int i = d >> 1;
  float freq = __expf(-(float)(2 * i) * (9.210340371976184f / (float)DD));
  float v = (float)s * freq;
  pe[idx] = (d & 1) ? __cosf(v) : __sinf(v);
}

// ---------------- generic bf16-WMMA GEMM, 128x128 block tile ----------------
// C(M,N) = A(M,K) @ Bm(K,N) [+ bias] [+ epilogue]
// EPI: 0 none, 1 +pos-encoding (extra = pe(S,N)), 2 gelu, 3 +residual (extra = (M,N))
// 256 threads = 8 waves; wave w computes rows 16w..16w+15 x all 128 cols.
template <int EPI>
__global__ void gemm_bf16_kernel(const float* __restrict__ A,
                                 const float* __restrict__ Bm,
                                 const float* __restrict__ bias,
                                 const float* __restrict__ extra,
                                 float* __restrict__ C,
                                 int M, int N, int K) {
  __shared__ unsigned short Abuf[128][40];   // [m][k] bf16, padded
  __shared__ unsigned short Bbuf[128][40];   // [n][k] bf16 (transposed), padded

  const int t    = threadIdx.x;             // 0..255
  const int lane = t & 31;
  const int w    = t >> 5;                  // 0..7
  const int hi   = lane >> 4;
  const int m    = lane & 15;
  const int n0   = blockIdx.x * 128;
  const int m0   = blockIdx.y * 128;

  v8f acc[8];
#pragma unroll
  for (int nt = 0; nt < 8; ++nt)
#pragma unroll
    for (int i = 0; i < 8; ++i) acc[nt][i] = 0.0f;

  for (int k0 = 0; k0 < K; k0 += 32) {
    __syncthreads();
    // stage A: 128x32, each thread converts 16 contiguous f32
    {
      const int row = t >> 1, ch = t & 1;
      const float* Ar = A + (size_t)(m0 + row) * K + k0 + ch * 16;
#pragma unroll
      for (int j = 0; j < 16; ++j) Abuf[row][ch * 16 + j] = f2bf(Ar[j]);
    }
    // stage B transposed: 32x128, each thread converts 16 contiguous f32
    {
      const int kr = t >> 3, ch = t & 7;
      const float* Br = Bm + (size_t)(k0 + kr) * N + n0 + ch * 16;
#pragma unroll
      for (int j = 0; j < 16; ++j) Bbuf[ch * 16 + j][kr] = f2bf(Br[j]);
    }
    __syncthreads();

    ABf a;
#pragma unroll
    for (int j = 0; j < 8; ++j) a.u[j]     = Abuf[16 * w + m][8 * hi + j];
#pragma unroll
    for (int j = 0; j < 8; ++j) a.u[8 + j] = Abuf[16 * w + m][8 * hi + 16 + j];

#pragma unroll
    for (int nt = 0; nt < 8; ++nt) {
      ABf b;
#pragma unroll
      for (int j = 0; j < 16; ++j) b.u[j] = Bbuf[nt * 16 + m][8 * hi + j + (j & 8)];
      acc[nt] = __builtin_amdgcn_wmma_f32_16x16x32_bf16(
          false, a.v, false, b.v, (short)0, acc[nt], false, false);
    }
  }

  // epilogue + writeback (C layout: lane col = m, VGPR i -> row i + 8*hi)
#pragma unroll
  for (int nt = 0; nt < 8; ++nt) {
#pragma unroll
    for (int i = 0; i < 8; ++i) {
      int row = m0 + 16 * w + i + 8 * hi;
      int col = n0 + nt * 16 + m;
      float vv = acc[nt][i];
      if (bias) vv += bias[col];
      if (EPI == 1) vv += extra[(size_t)(row & (SS - 1)) * N + col];
      if (EPI == 2) vv = gelu_exact(vv);
      if (EPI == 3) vv += extra[(size_t)row * N + col];
      C[(size_t)row * N + col] = vv;
    }
  }
}

// ---------------- attention: av = (scale*Q K^T, masked) @ V ----------------
// (faithful quirk: raw masked scores times V, no softmax)
// 256 threads = 8 waves; q-tile = 128 rows; keys streamed 32 at a time.
// Mask is applied while assembling the A-operand of the S@V WMMA, where each
// lane's mask bytes (8 consecutive keys of its own q-row) are contiguous.
__global__ void attn_kernel(const float* __restrict__ q,
                            const float* __restrict__ k,
                            const float* __restrict__ v,
                            const unsigned char* __restrict__ mask,
                            float* __restrict__ av) {
  __shared__ unsigned short Kbuf[32][72];      // [key][d]
  __shared__ unsigned short Vbuf[64][40];      // [d][key] (transposed)
  __shared__ unsigned short Sbuf[8][16][40];   // per wave: [qrow][key] (unmasked)
  __shared__ unsigned char  Mbuf[128][32];     // mask tile [qrow][key]

  const int b  = blockIdx.z;
  const int h  = blockIdx.y;
  const int q0 = blockIdx.x * 128;
  const int t    = threadIdx.x;
  const int lane = t & 31;
  const int w    = t >> 5;                     // 0..7
  const int hi   = lane >> 4;
  const int m    = lane & 15;
  const float scale = 0.125f;   // 1/sqrt(64)

  // Q strip (16x64 per wave) as two A-operands (d=0..31, 32..63), pre-scaled
  ABf qa[2];
  {
    const float* qr = q + ((size_t)(b * SS) + q0 + 16 * w + m) * DD + h * DHH;
#pragma unroll
    for (int c = 0; c < 2; ++c) {
#pragma unroll
      for (int j = 0; j < 8; ++j)
        qa[c].u[j]     = f2bf(qr[32 * c + 8 * hi + j] * scale);
#pragma unroll
      for (int j = 0; j < 8; ++j)
        qa[c].u[8 + j] = f2bf(qr[32 * c + 8 * hi + 16 + j] * scale);
    }
  }

  v8f acc[4];
#pragma unroll
  for (int nt = 0; nt < 4; ++nt)
#pragma unroll
    for (int i = 0; i < 8; ++i) acc[nt][i] = 0.0f;

  for (int k0 = 0; k0 < SS; k0 += 32) {
    __syncthreads();
    // stage K (32x64) and V (transposed) tiles: each thread converts 8 f32 each
    {
      const int row = t >> 3, ch = t & 7;
      const float* kr = k + ((size_t)(b * SS) + k0 + row) * DD + h * DHH + ch * 8;
      const float* vr = v + ((size_t)(b * SS) + k0 + row) * DD + h * DHH + ch * 8;
#pragma unroll
      for (int j = 0; j < 8; ++j) Kbuf[row][ch * 8 + j] = f2bf(kr[j]);
#pragma unroll
      for (int j = 0; j < 8; ++j) Vbuf[ch * 8 + j][row] = f2bf(vr[j]);
    }
    // stage mask tile 128x32 bytes via CDNA5 async global->LDS copy (16B/lane)
    {
      const int qlocal = t >> 1, half = t & 1;
      const unsigned char* mp =
          mask + ((size_t)b * SS + q0 + qlocal) * SS + k0 + 16 * half;
      unsigned lds_off = (unsigned)(size_t)&Mbuf[qlocal][16 * half];
      async_copy_b128(lds_off, (const void*)mp);
    }
    wait_async0();
    __syncthreads();

    // two 16-key score tiles (unmasked; stashed to per-wave LDS in A-layout)
#pragma unroll
    for (int kt = 0; kt < 2; ++kt) {
      v8f st;
#pragma unroll
      for (int i = 0; i < 8; ++i) st[i] = 0.0f;
#pragma unroll
      for (int c = 0; c < 2; ++c) {
        ABf bb;                       // B-operand: col = key, K = d-chunk
#pragma unroll
        for (int j = 0; j < 16; ++j)
          bb.u[j] = Kbuf[16 * kt + m][32 * c + 8 * hi + j + (j & 8)];
        st = __builtin_amdgcn_wmma_f32_16x16x32_bf16(
            false, qa[c].v, false, bb.v, (short)0, st, false, false);
      }
#pragma unroll
      for (int i = 0; i < 8; ++i)
        Sbuf[w][i + 8 * hi][16 * kt + m] = f2bf(st[i]);
    }

    // Assemble masked A-operand: lane = q-row (16w+m), keys 8hi+{0..7,16..23}.
    // Mask bytes for those keys are two contiguous aligned 8B chunks.
    ABf sa;
#pragma unroll
    for (int j = 0; j < 8; ++j) sa.u[j]     = Sbuf[w][m][8 * hi + j];
#pragma unroll
    for (int j = 0; j < 8; ++j) sa.u[8 + j] = Sbuf[w][m][8 * hi + 16 + j];
    {
      const unsigned long long mk0 =
          *(const unsigned long long*)&Mbuf[16 * w + m][8 * hi];
      const unsigned long long mk1 =
          *(const unsigned long long*)&Mbuf[16 * w + m][8 * hi + 16];
#pragma unroll
      for (int j = 0; j < 8; ++j)
        if (((mk0 >> (8 * j)) & 0xFFull) == 0) sa.u[j] = (unsigned short)BF16_NEG1E9;
#pragma unroll
      for (int j = 0; j < 8; ++j)
        if (((mk1 >> (8 * j)) & 0xFFull) == 0) sa.u[8 + j] = (unsigned short)BF16_NEG1E9;
    }

    // AV accumulation: A = masked scores (16x32 keys), B = Vbuf (32 x 64 d)
#pragma unroll
    for (int nt = 0; nt < 4; ++nt) {
      ABf vbf;
#pragma unroll
      for (int j = 0; j < 16; ++j) vbf.u[j] = Vbuf[nt * 16 + m][8 * hi + j + (j & 8)];
      acc[nt] = __builtin_amdgcn_wmma_f32_16x16x32_bf16(
          false, sa.v, false, vbf.v, (short)0, acc[nt], false, false);
    }
  }

#pragma unroll
  for (int nt = 0; nt < 4; ++nt)
#pragma unroll
    for (int i = 0; i < 8; ++i)
      av[((size_t)(b * SS) + q0 + 16 * w + i + 8 * hi) * DD + h * DHH + nt * 16 + m] =
          acc[nt][i];
}

// ---------------- LayerNorm (one row per block, 256 threads, D=512) ----------------
__global__ void ln_kernel(const float* __restrict__ in,
                          const float* __restrict__ g,
                          const float* __restrict__ bta,
                          float* __restrict__ out) {
  __shared__ float red[256];
  const int row = blockIdx.x, t = threadIdx.x;
  const float* r = in + (size_t)row * DD;
  float x0 = r[t], x1 = r[t + 256];
  red[t] = x0 + x1;
  __syncthreads();
  for (int s = 128; s > 0; s >>= 1) {
    if (t < s) red[t] += red[t + s];
    __syncthreads();
  }
  float mean = red[0] * (1.0f / DD);
  __syncthreads();
  float d0 = x0 - mean, d1 = x1 - mean;
  red[t] = d0 * d0 + d1 * d1;
  __syncthreads();
  for (int s = 128; s > 0; s >>= 1) {
    if (t < s) red[t] += red[t + s];
    __syncthreads();
  }
  float inv = rsqrtf(red[0] * (1.0f / DD) + 1e-5f);
  out[(size_t)row * DD + t]       = d0 * inv * g[t]       + bta[t];
  out[(size_t)row * DD + t + 256] = d1 * inv * g[t + 256] + bta[t + 256];
}

// ---------------- prediction head: (8, S*D) @ (S*D, 64) + b ----------------
__global__ void pred_kernel(const float* __restrict__ x,
                            const float* __restrict__ wpred,
                            const float* __restrict__ bpred,
                            float* __restrict__ out) {
  __shared__ float red[256];
  const int n = blockIdx.x & 63;
  const int b = blockIdx.x >> 6;
  const int t = threadIdx.x;
  const size_t K = (size_t)SS * DD;   // 524288
  const float* xr = x + (size_t)b * K;
  float s = 0.0f;
  for (size_t i = t; i < K; i += 256) s += xr[i] * wpred[i * DATA + n];
  red[t] = s;
  __syncthreads();
  for (int st = 128; st > 0; st >>= 1) {
    if (t < st) red[t] += red[t + st];
    __syncthreads();
  }
  if (t == 0) out[b * DATA + n] = red[0] + bpred[n];
}

// ---------------- host-side launcher ----------------
extern "C" void kernel_launch(void* const* d_in, const int* in_sizes, int n_in,
                              void* d_out, int out_size, void* d_ws, size_t ws_size,
                              hipStream_t stream) {
  const float* data    = (const float*)d_in[0];
  const unsigned char* mask = (const unsigned char*)d_in[1];
  const float* embed_w = (const float*)d_in[2];
  const float* embed_b = (const float*)d_in[3];
  const float* wq      = (const float*)d_in[4];
  const float* wk      = (const float*)d_in[5];
  const float* wv      = (const float*)d_in[6];
  const float* wo      = (const float*)d_in[7];
  const float* wo_b    = (const float*)d_in[8];
  const float* ln1_g   = (const float*)d_in[9];
  const float* ln1_b   = (const float*)d_in[10];
  const float* w1      = (const float*)d_in[11];
  const float* b1      = (const float*)d_in[12];
  const float* w2      = (const float*)d_in[13];
  const float* b2      = (const float*)d_in[14];
  const float* ln2_g   = (const float*)d_in[15];
  const float* ln2_b   = (const float*)d_in[16];
  const float* pred_w  = (const float*)d_in[17];
  const float* pred_b  = (const float*)d_in[18];
  float* outp = (float*)d_out;

  // workspace layout
  char* wsp = (char*)d_ws;
  float* pe  = (float*)wsp;  wsp += (size_t)SS * DD * 4;
  float* x   = (float*)wsp;  wsp += (size_t)MM * DD * 4;
  float* tmp = (float*)wsp;  wsp += (size_t)MM * DD * 4;
  float* qb  = (float*)wsp;  wsp += (size_t)MM * DD * 4;
  float* kb  = (float*)wsp;  wsp += (size_t)MM * DD * 4;
  float* vb  = (float*)wsp;  wsp += (size_t)MM * DD * 4;
  float* av  = (float*)wsp;  wsp += (size_t)MM * DD * 4;
  float* hb  = (float*)wsp;  wsp += (size_t)MM * DFF * 4;

  // 1) positional encoding
  pe_kernel<<<(SS * DD) / 256, 256, 0, stream>>>(pe);

  // 2) embed: x = data @ embed_w + embed_b + PE
  gemm_bf16_kernel<1><<<dim3(DD / 128, MM / 128), 256, 0, stream>>>(
      data, embed_w, embed_b, pe, x, MM, DD, DATA);

  for (int l = 0; l < LL; ++l) {
    const size_t oDD = (size_t)l * DD * DD;
    // QKV projections (no bias in reference)
    gemm_bf16_kernel<0><<<dim3(DD / 128, MM / 128), 256, 0, stream>>>(
        x, wq + oDD, nullptr, nullptr, qb, MM, DD, DD);
    gemm_bf16_kernel<0><<<dim3(DD / 128, MM / 128), 256, 0, stream>>>(
        x, wk + oDD, nullptr, nullptr, kb, MM, DD, DD);
    gemm_bf16_kernel<0><<<dim3(DD / 128, MM / 128), 256, 0, stream>>>(
        x, wv + oDD, nullptr, nullptr, vb, MM, DD, DD);

    // attention (raw masked scores @ V)
    attn_kernel<<<dim3(SS / 128, HH, BB), 256, 0, stream>>>(qb, kb, vb, mask, av);

    // output projection + residual, then LN1
    gemm_bf16_kernel<3><<<dim3(DD / 128, MM / 128), 256, 0, stream>>>(
        av, wo + oDD, wo_b + (size_t)l * DD, x, tmp, MM, DD, DD);
    ln_kernel<<<MM, 256, 0, stream>>>(tmp, ln1_g + (size_t)l * DD,
                                      ln1_b + (size_t)l * DD, x);

    // FFN
    gemm_bf16_kernel<2><<<dim3(DFF / 128, MM / 128), 256, 0, stream>>>(
        x, w1 + (size_t)l * DD * DFF, b1 + (size_t)l * DFF, nullptr, hb,
        MM, DFF, DD);
    gemm_bf16_kernel<3><<<dim3(DD / 128, MM / 128), 256, 0, stream>>>(
        hb, w2 + (size_t)l * DFF * DD, b2 + (size_t)l * DD, x, tmp,
        MM, DD, DFF);
    ln_kernel<<<MM, 256, 0, stream>>>(tmp, ln2_g + (size_t)l * DD,
                                      ln2_b + (size_t)l * DD, x);
  }

  // prediction head
  pred_kernel<<<BB * DATA, 256, 0, stream>>>(x, pred_w, pred_b, outp);
}
